// WaveOperator_24120536335071
// MI455X (gfx1250) — matrix operand order repaired and checked by reading, not compiled, not run on hardware
//
#include <hip/hip_runtime.h>

typedef _Float16 v16h __attribute__((ext_vector_type(16)));
typedef _Float16 v8h  __attribute__((ext_vector_type(8)));
typedef _Float16 v4h  __attribute__((ext_vector_type(4)));
typedef float    v8f  __attribute__((ext_vector_type(8)));

#define B_    16
#define NIN_  256
#define N_    512            // Nout
#define P_    256            // Nphi
#define T_    240            // Nt
#define K_    1024           // 2*N_ (re|im along K)
#define M_    (B_*N_)        // 8192 GEMM rows
#define NC_   (2*P_)         // 512 GEMM cols (Zr|Zi)
#define TWO_PI 6.283185307179586f

// ---------------- prep: shared twiddle tables cos/sin(2*pi*m/512) ----------------
__global__ void k_tables(float* __restrict__ ct, float* __restrict__ st) {
  int m = blockIdx.x * 256 + threadIdx.x;          // 512 total
  float a = TWO_PI * (float)m / (float)N_;
  ct[m] = cosf(a);
  st[m] = sinf(a);
}

// ---------------- prep: pack sampling trig matrix into WMMA B-fragment layout ----
// B'[k',n'] : k' = slot*512+ky (slot0=Re rows, slot1=Im rows)
//             n' = p (Zr col) or 256+p (Zi col)
// Zr col: [cos ; -sin], Zi col: [sin ; cos], angle = 2*pi*ky*idx1[p]/512
// Packed so a lane's 16 halves (one v16h) are contiguous 32B.
__global__ void k_bpack(const int* __restrict__ idx1, _Float16* __restrict__ bp) {
  int gid  = blockIdx.x * 256 + threadIdx.x;       // 32nt*32kt*32lane*16e = 524288
  int e    =  gid        & 15;
  int lane = (gid >> 4)  & 31;
  int kt   = (gid >> 9)  & 31;
  int nt   =  gid >> 14;
  int kc   = (e < 8) ? ((lane >> 4) * 8 + e) : (16 + (lane >> 4) * 8 + (e - 8)); // B frag: lanes 0-15 K={0..7,16..23}, 16-31 K={8..15,24..31}
  int kp   = kt * 32 + kc;
  int ky   = kp & (N_ - 1);
  int slot = kp >> 9;
  int nc   = nt * 16 + (lane & 15);
  int p    = nc & (P_ - 1);
  int zi   = nc >> 8;
  int m    = (ky * idx1[p]) & (N_ - 1);
  float ang = TWO_PI * (float)m / (float)N_;
  float c = cosf(ang), s = sinf(ang);
  float v = zi ? (slot ? c : s) : (slot ? -s : c);
  bp[gid] = (_Float16)v;
}

// ---------------- one-time forward DFT along y: U[b,x,{re,im},ky] -----------------
// U = sum_{y=0..255} f[b,x,y] * exp(-i*2pi*ky*(y+128)/512)  (zero padding implicit)
__global__ void k_dft_y(const float* __restrict__ f, const float* __restrict__ ct,
                        const float* __restrict__ st, float* __restrict__ U) {
  __shared__ float lc[N_], ls[N_], fr[NIN_];
  int t = threadIdx.x;
  lc[t] = ct[t]; lc[t + 256] = ct[t + 256];
  ls[t] = st[t]; ls[t + 256] = st[t + 256];
  int b = blockIdx.x >> 8, x = blockIdx.x & 255;
  fr[t] = f[(size_t)(b * NIN_ + x) * NIN_ + t];
  __syncthreads();
  for (int q = 0; q < 2; ++q) {
    int ky = t + q * 256;
    int m = (ky * 128) & (N_ - 1);                 // y starts at 128
    float ur = 0.f, ui = 0.f;
#pragma unroll 4
    for (int y = 0; y < 256; ++y) {
      float fv = fr[y];
      ur = fmaf(fv, lc[m], ur);
      ui = fmaf(-fv, ls[m], ui);
      m = (m + ky) & (N_ - 1);
    }
    float* ub = U + (size_t)(b * 256 + x) * 2 * N_;
    ub[ky] = ur; ub[N_ + ky] = ui;
  }
}

// ---------------- one-time forward DFT along x: state S[b,kx,{re,im},ky] ----------
__global__ void k_dft_x(const float* __restrict__ U, const float* __restrict__ ct,
                        const float* __restrict__ st, float* __restrict__ S) {
  __shared__ float lc[N_], ls[N_];
  int t = threadIdx.x;
  lc[t] = ct[t]; lc[t + 256] = ct[t + 256];
  ls[t] = st[t]; ls[t + 256] = st[t + 256];
  __syncthreads();
  int b = blockIdx.x >> 9, kx = blockIdx.x & (N_ - 1);
  for (int q = 0; q < 2; ++q) {
    int ky = t + q * 256;
    float frv = 0.f, fiv = 0.f;
    int m = (kx * 128) & (N_ - 1);                 // x starts at 128
    for (int x = 0; x < 256; ++x) {
      const float* ub = U + (size_t)(b * 256 + x) * 2 * N_;
      float ur = ub[ky], ui = ub[N_ + ky];
      float c = lc[m], s = ls[m];
      frv = fmaf(c, ur, fmaf( s, ui, frv));        // (c - i s)(ur + i ui)
      fiv = fmaf(c, ui, fmaf(-s, ur, fiv));
      m = (m + kx) & (N_ - 1);
    }
    float* sb = S + (size_t)(b * N_ + kx) * 2 * N_;
    sb[ky] = frv; sb[N_ + ky] = fiv;
  }
}

// ---------------- per-step spectral recurrence (L2-resident, elementwise) ---------
// Wn = (2 - 4*sin_filter)*W - Walt ; also emit f16 copy for the WMMA GEMM
__global__ void k_evolve(const float* W, const float* Wa, float* Wn,
                         _Float16* A16, const float* __restrict__ sinfil) {
  size_t i4 = (size_t)blockIdx.x * 256 + threadIdx.x;  // 2,097,152 float4's
  size_t i  = i4 * 4;
  int ky = (int)(i & (N_ - 1));
  int kx = (int)((i >> 10) & (N_ - 1));
  const float4 sf = *(const float4*)(sinfil + (size_t)kx * N_ + ky);
  const float4 w  = ((const float4*)W)[i4];
  const float4 wa = ((const float4*)Wa)[i4];
  float4 wn;
  wn.x = fmaf(2.f - 4.f * sf.x, w.x, -wa.x);
  wn.y = fmaf(2.f - 4.f * sf.y, w.y, -wa.y);
  wn.z = fmaf(2.f - 4.f * sf.z, w.z, -wa.z);
  wn.w = fmaf(2.f - 4.f * sf.w, w.w, -wa.w);
  ((float4*)Wn)[i4] = wn;
  v4h h = { (_Float16)wn.x, (_Float16)wn.y, (_Float16)wn.z, (_Float16)wn.w };
  *(v4h*)(A16 + i) = h;
}

// ---------------- per-step sampling GEMM: Zt = A16(8192x1024) * Bp(1024x512) ------
// f16 x f16 -> f32 via v_wmma_f32_16x16x32_f16, register-tiled: each wave computes a
// 64x64 output block (4x4 WMMA tiles), so each A/B fragment load feeds 4 WMMAs.
// Z stored transposed [col][row] so the sensor reduction reads coalesced and each
// lane stores 8 contiguous rows (two b128 stores per tile).
__global__ void __launch_bounds__(128)
k_gemm(const _Float16* __restrict__ A, const _Float16* __restrict__ Bp,
       float* __restrict__ Zt) {
  int lane = threadIdx.x & 31;
  int wv   = threadIdx.x >> 5;
  int mg   = blockIdx.x;                        // 128 groups of 64 rows
  int ng   = blockIdx.y * 4 + wv;               // 8  groups of 64 cols
  int r16  = lane & 15;
  int hi   = lane >> 4;                         // half-wave select
  // A frag (16x32 f16): lanes 0-15 row M=r16 K={0..7,16..23}; lanes 16-31 K={8..15,24..31}
  const _Float16* pa = A + (size_t)(mg * 64 + r16) * K_ + hi * 8;
  const _Float16* pb = Bp + ((size_t)(ng * 4) * 32 * 32 + lane) * 16;
  v8f acc[4][4];
#pragma unroll
  for (int im = 0; im < 4; ++im)
#pragma unroll
    for (int in = 0; in < 4; ++in)
      acc[im][in] = (v8f){0.f, 0.f, 0.f, 0.f, 0.f, 0.f, 0.f, 0.f};

  for (int kt = 0; kt < 32; ++kt) {
    v16h av[4], bv[4];
#pragma unroll
    for (int im = 0; im < 4; ++im) {
      const _Float16* p = pa + (size_t)im * 16 * K_;
      v8h a0 = *(const v8h*)(p);
      v8h a1 = *(const v8h*)(p + 16);
      av[im] = __builtin_shufflevector(a0, a1, 0,1,2,3,4,5,6,7,8,9,10,11,12,13,14,15);
    }
#pragma unroll
    for (int in = 0; in < 4; ++in)
      bv[in] = *(const v16h*)(pb + (size_t)in * 32 * 32 * 16);
#pragma unroll
    for (int im = 0; im < 4; ++im)
#pragma unroll
      for (int in = 0; in < 4; ++in)
        acc[im][in] = __builtin_amdgcn_wmma_f32_16x16x32_f16(
            false, av[im], false, bv[in], (short)0, acc[im][in], false, false);
    pa += 32; pb += 32 * 16;
    __builtin_prefetch(pa, 0, 3);               // keep the A stream warm near the WGP
  }
  // C frag (f32 16x16): VGPR r -> row r (lanes 0-15) / row 8+r (lanes 16-31), col = lane%16
#pragma unroll
  for (int im = 0; im < 4; ++im)
#pragma unroll
    for (int in = 0; in < 4; ++in) {
      int col  = (ng * 4 + in) * 16 + r16;
      int row0 = mg * 64 + im * 16 + hi * 8;
      float4 lo = { acc[im][in][0], acc[im][in][1], acc[im][in][2], acc[im][in][3] };
      float4 hg = { acc[im][in][4], acc[im][in][5], acc[im][in][6], acc[im][in][7] };
      float* pz = Zt + (size_t)col * M_ + row0;
      *(float4*)(pz)     = lo;
      *(float4*)(pz + 4) = hg;
    }
}

// ---------------- per-step sensor reduce over kx: y[b,p,t] ------------------------
__global__ void k_stage2(const float* __restrict__ Zt, const int* __restrict__ idx0,
                         const float* __restrict__ ct, const float* __restrict__ st,
                         float* __restrict__ out, int t) {
  __shared__ float lc[N_], ls[N_], red[256];
  int tid = threadIdx.x;
  lc[tid] = ct[tid]; lc[tid + 256] = ct[tid + 256];
  ls[tid] = st[tid]; ls[tid + 256] = st[tid + 256];
  __syncthreads();
  int p = blockIdx.x & (P_ - 1);
  int b = blockIdx.x >> 8;
  int xp = idx0[p];
  float sum = 0.f;
  for (int q = 0; q < 2; ++q) {
    int kx = tid + q * 256;
    int row = b * N_ + kx;
    int m = (kx * xp) & (N_ - 1);
    float zr = Zt[(size_t)p * M_ + row];
    float zi = Zt[(size_t)(P_ + p) * M_ + row];
    sum = fmaf(zr, lc[m], fmaf(-zi, ls[m], sum)); // Re(Z * e^{+i a kx})
  }
  red[tid] = sum;
  __syncthreads();
  for (int s = 128; s > 0; s >>= 1) {
    if (tid < s) red[tid] += red[tid + s];
    __syncthreads();
  }
  if (tid == 0)
    out[((size_t)b * P_ + p) * T_ + t] = red[0] * (1.0f / 262144.0f); // 1/N^2 (ifft)
}

extern "C" void kernel_launch(void* const* d_in, const int* in_sizes, int n_in,
                              void* d_out, int out_size, void* d_ws, size_t ws_size,
                              hipStream_t stream) {
  const float* f      = (const float*)d_in[0];   // [16,1,256,256]
  const float* sinfil = (const float*)d_in[1];   // [512,512]
  const int*   idx0   = (const int*)d_in[2];     // [256]
  const int*   idx1   = (const int*)d_in[3];     // [256]
  (void)in_sizes; (void)n_in; (void)out_size;    // Nt fixed to 240

  char* w = (char*)d_ws;
  float*    S0  = (float*)w;    w += (size_t)M_ * K_ * 4;   // 33.5 MB spectral state A
  float*    S1  = (float*)w;    w += (size_t)M_ * K_ * 4;   // 33.5 MB spectral state B
  _Float16* A16 = (_Float16*)w; w += (size_t)M_ * K_ * 2;   // 16.8 MB f16 GEMM operand
  float*    Zt  = (float*)w;    w += (size_t)NC_ * M_ * 4;  // 16.8 MB Z (also DFT temp U)
  _Float16* Bp  = (_Float16*)w; w += (size_t)K_ * NC_ * 2;  // 1.0 MB packed trig B
  float*    ct  = (float*)w;    w += N_ * 4;
  float*    st  = (float*)w;    w += N_ * 4;
  (void)ws_size;
  float* U = Zt;  // forward-DFT intermediate reuses Z scratch (loop hasn't started)

  k_tables<<<2, 256, 0, stream>>>(ct, st);
  k_bpack <<<2048, 256, 0, stream>>>(idx1, Bp);
  k_dft_y <<<B_ * NIN_, 256, 0, stream>>>(f, ct, st, U);
  k_dft_x <<<B_ * N_,  256, 0, stream>>>(U, ct, st, S0);

  const float* Wc = S0;   // current W  (spectral)
  const float* Wa = S0;   // previous W (== W at t=0, matching (W0, W0) carry)
  float*       Wo = S1;   // where Wn is written
  dim3 ggrid(M_ / 64, NC_ / 64 / 4);              // (128, 2), 4 waves per block
  for (int t = 0; t < T_; ++t) {
    k_evolve<<<8192, 256, 0, stream>>>(Wc, Wa, Wo, A16, sinfil);
    k_gemm  <<<ggrid, 128, 0, stream>>>(A16, Bp, Zt);
    k_stage2<<<B_ * P_, 256, 0, stream>>>(Zt, idx0, ct, st, (float*)d_out, t);
    const float* oldW = Wc;
    Wc = Wo; Wa = oldW; Wo = (float*)oldW;   // ping-pong (Wa==Wo aliasing is per-element safe)
  }
}
